// selfAttn2d_80882824118316
// MI455X (gfx1250) — matrix operand order repaired and checked by reading, not compiled, not run on hardware
//
#include <hip/hip_runtime.h>
#include <math.h>

typedef __attribute__((ext_vector_type(16))) _Float16 v16h;
typedef __attribute__((ext_vector_type(8)))  float    v8f;

#define DEVINL __device__ __forceinline__

namespace {

// ---------------- shared memory layout (units: _Float16 halves) ----------------
constexpr int H_WQ    = 0;                    // 64x64
constexpr int H_WK    = H_WQ   + 64*64;
constexpr int H_WV    = H_WK   + 64*64;
constexpr int H_WREL  = H_WV   + 64*64;       // [9][64][64]
constexpr int H_WDT   = H_WREL + 9*64*64;     // w_deconv transposed: [o*16+s][ch] = [1024][64]
constexpr int H_WPROJ = H_WDT  + 1024*64;     // [o][c]
constexpr int H_WSF1  = H_WPROJ + 64*64;      // [16][32]
constexpr int H_WSF2  = H_WSF1 + 16*32;       // [16][16]
constexpr int H_ZROW  = H_WSF2 + 16*16;       // 64 zeros
constexpr int H_WAVE  = H_ZROW + 64;

// per-wave scratch (stride WAVE_H halves)
constexpr int PW_SPT   = 0;                   // spT[p][c]    16x64
constexpr int PW_QF    = PW_SPT   + 16*64;    // qf[o][p]     64x16  (scaled by 1/4)
constexpr int PW_KF    = PW_QF    + 64*16;    // kf[o][p]
constexpr int PW_VMAT  = PW_KF    + 64*16;    // v[o][p]
constexpr int PW_RELCT = PW_VMAT  + 64*16;    // rel_cT[p][ch] 16x64
constexpr int PW_CATB  = PW_RELCT + 16*64;    // cat[p][t]    16x32
constexpr int PW_H1B   = PW_CATB  + 16*32;    // h1[p][m]     16x16
constexpr int PW_WTILE = PW_H1B   + 16*16;    // weights[p][s] 16x16
constexpr int WAVE_H   = PW_WTILE + 16*16;    // 6144 halves

constexpr int H_TOTAL  = H_WAVE + 4*WAVE_H;
constexpr int F_BASE_B = ((H_TOTAL*2 + 15)/16)*16;
// float region (units: floats, offset from F_BASE_B)
constexpr int F_BSF1  = 0;     // 16
constexpr int F_BSF2  = 16;    // 16
constexpr int F_BPROJ = 32;    // 64
constexpr int F_SVEC  = 96;    // 4 waves x 64
constexpr int F_TOTAL = F_SVEC + 4*64;
constexpr size_t SMEM_BYTES = (size_t)F_BASE_B + (size_t)F_TOTAL*4;

constexpr int PATCHES_PER_WAVE = 4;   // 256 blocks * 4 waves * 4 = 4096 patches

} // namespace

union FragU { float4 f4[2]; v16h h; };

// A-fragment (16x32, f16) from row-major A; `row` = base of this lane's row (m = lane&15)
DEVINL v16h ldsA32(const _Float16* row, int k0, int hi) {
  FragU u;
  u.f4[0] = *(const float4*)(row + k0 + 8*hi);
  u.f4[1] = *(const float4*)(row + k0 + 16 + 8*hi);
  return u.h;
}
// A-fragment with K=16 zero-padded to 32 (elements 8..15 are K>=16 -> zero)
DEVINL v16h ldsA16(const _Float16* row, int hi) {
  FragU u;
  u.f4[0] = *(const float4*)(row + 8*hi);
  u.f4[1] = make_float4(0.f, 0.f, 0.f, 0.f);
  return u.h;
}
// B-fragment (32x16, f16) from N-major (transposed) B; `row` = base of column n = lane&15
DEVINL v16h ldsB32(const _Float16* row, int k0, int kh) {
  const _Float16* p = row + k0 + 8*kh;
  FragU u;
  u.f4[0] = *(const float4*)(p);
  u.f4[1] = *(const float4*)(p + 16);
  return u.h;
}
// B-fragment with K=16 zero-padded to 32 (elements 8..15 are K>=16 -> zero)
DEVINL v16h ldsB16(const _Float16* row, int kh, const _Float16* zrow) {
  const _Float16* p = row + 8*kh;
  FragU u;
  u.f4[0] = *(const float4*)(p);
  u.f4[1] = make_float4(0.f, 0.f, 0.f, 0.f);
  return u.h;
}

DEVINL v8f wmma16(v16h a, v16h b, v8f c) {
  return __builtin_amdgcn_wmma_f32_16x16x32_f16(false, a, false, b, (short)0, c, false, false);
}

__global__ __launch_bounds__(128, 1) void selfattn2d_fused(
    const float* __restrict__ x,
    const float* __restrict__ gwq,   const float* __restrict__ gwk,
    const float* __restrict__ gwv,   const float* __restrict__ gwrel,
    const float* __restrict__ gwdec, const float* __restrict__ gwsf1,
    const float* __restrict__ gbsf1, const float* __restrict__ gwsf2,
    const float* __restrict__ gbsf2, const float* __restrict__ gwproj,
    const float* __restrict__ gbproj, float* __restrict__ out)
{
  extern __shared__ char smem_raw[];
  _Float16* hs = (_Float16*)smem_raw;
  float*    fs = (float*)(smem_raw + F_BASE_B);

  const int tid  = threadIdx.x;
  const int wid  = tid >> 5;
  const int lane = tid & 31;
  const int m16  = lane & 15;
  const int hi   = lane >> 4;

  // ---------------- stage weights (f32 -> f16) once per block ----------------
  for (int i = tid; i < 64*64; i += 128) {
    hs[H_WQ    + i] = (_Float16)gwq[i];
    hs[H_WK    + i] = (_Float16)gwk[i];
    hs[H_WV    + i] = (_Float16)gwv[i];
    hs[H_WPROJ + i] = (_Float16)gwproj[i];
  }
  for (int i = tid; i < 9*64*64; i += 128) {   // i enumerates src order (coalesced read)
    int o = i / 576, rem = i % 576, c = rem / 9, uv = rem % 9;
    hs[H_WREL + uv*4096 + o*64 + c] = (_Float16)gwrel[i];
  }
  for (int i = tid; i < 1024*64; i += 128) {   // wdT[col][ch] = wdec[ch*1024 + col]
    int ch = i >> 10, col = i & 1023;
    hs[H_WDT + col*64 + ch] = (_Float16)gwdec[i];
  }
  for (int i = tid; i < 16*32; i += 128) hs[H_WSF1 + i] = (_Float16)gwsf1[i];
  for (int i = tid; i < 16*16; i += 128) hs[H_WSF2 + i] = (_Float16)gwsf2[i];
  if (tid < 64) hs[H_ZROW + tid] = (_Float16)0.f;
  if (tid < 16) { fs[F_BSF1 + tid] = gbsf1[tid]; fs[F_BSF2 + tid] = gbsf2[tid]; }
  if (tid < 64) fs[F_BPROJ + tid] = gbproj[tid];
  __syncthreads();

  const _Float16* wq    = hs + H_WQ;
  const _Float16* wk    = hs + H_WK;
  const _Float16* wvv   = hs + H_WV;
  const _Float16* wrel  = hs + H_WREL;
  const _Float16* wdt   = hs + H_WDT;
  const _Float16* wproj = hs + H_WPROJ;
  const _Float16* wsf1  = hs + H_WSF1;
  const _Float16* wsf2  = hs + H_WSF2;
  const _Float16* zrow  = hs + H_ZROW;

  _Float16* spT   = hs + H_WAVE + wid*WAVE_H + PW_SPT;
  _Float16* qf    = hs + H_WAVE + wid*WAVE_H + PW_QF;
  _Float16* kf    = hs + H_WAVE + wid*WAVE_H + PW_KF;
  _Float16* vmat  = hs + H_WAVE + wid*WAVE_H + PW_VMAT;
  _Float16* relcT = hs + H_WAVE + wid*WAVE_H + PW_RELCT;
  _Float16* catb  = hs + H_WAVE + wid*WAVE_H + PW_CATB;
  _Float16* h1b   = hs + H_WAVE + wid*WAVE_H + PW_H1B;
  _Float16* wtile = hs + H_WAVE + wid*WAVE_H + PW_WTILE;
  float*    svec  = fs + F_SVEC + wid*64;

  for (int it = 0; it < PATCHES_PER_WAVE; ++it) {
    const int pid  = blockIdx.x * (4*PATCHES_PER_WAVE) + it*4 + wid;
    const int bimg = pid >> 10;
    const int rem  = pid & 1023;
    const int ho   = rem >> 5;
    const int wo   = rem & 31;

    // -------- gather patch: spT[p][c], zero-padded unfold (stride 2, pad 1) --------
    for (int p = 0; p < 16; ++p) {                    // uniform per wave
      const int ii = ho*2 - 1 + (p >> 2);
      const int jj = wo*2 - 1 + (p & 3);
      if ((unsigned)ii < 64u && (unsigned)jj < 64u) {
        const float* src = x + (((size_t)bimg*64 + lane)*64 + ii)*64 + jj;
        spT[p*64 + lane]      = (_Float16)src[0];
        spT[p*64 + lane + 32] = (_Float16)src[32*64*64];
      } else {
        spT[p*64 + lane]      = (_Float16)0.f;
        spT[p*64 + lane + 32] = (_Float16)0.f;
      }
    }

    // -------- QKV projections: D[o-tile, p] = W(16x32 tiles) x spT --------
    {
      const _Float16* brow = spT + m16*64;
      const v16h b0 = ldsB32(brow, 0,  hi);    // spT fragments invariant across mt
      const v16h b1 = ldsB32(brow, 32, hi);
      for (int mt = 0; mt < 4; ++mt) {
        const _Float16* arQ = wq  + (16*mt + m16)*64;
        const _Float16* arK = wk  + (16*mt + m16)*64;
        const _Float16* arV = wvv + (16*mt + m16)*64;
        v8f aq = {}, ak = {}, av = {};
        aq = wmma16(ldsA32(arQ, 0,  hi), b0, aq);
        aq = wmma16(ldsA32(arQ, 32, hi), b1, aq);
        ak = wmma16(ldsA32(arK, 0,  hi), b0, ak);
        ak = wmma16(ldsA32(arK, 32, hi), b1, ak);
        av = wmma16(ldsA32(arV, 0,  hi), b0, av);
        av = wmma16(ldsA32(arV, 32, hi), b1, av);
        #pragma unroll
        for (int r = 0; r < 8; ++r) {
          const int o = 16*mt + r + 8*hi;
          qf[o*16 + m16]   = (_Float16)(aq[r] * 0.25f);   // DKH^-0.5 = 1/4
          kf[o*16 + m16]   = (_Float16)ak[r];
          vmat[o*16 + m16] = (_Float16)av[r];
        }
      }
    }

    // -------- 3x3 conv on the 4x4 patch: 9 shifted GEMMs -> rel_cT[p][ch] --------
    {
      const int pi = m16 >> 2, pj = m16 & 3;
      for (int mt = 0; mt < 4; ++mt) {
        v8f acc = {};
        #pragma unroll
        for (int u = 0; u < 3; ++u)
          #pragma unroll
          for (int v2 = 0; v2 < 3; ++v2) {
            const int si = pi + u - 1, sj = pj + v2 - 1;
            const _Float16* brow =
                ((unsigned)si < 4u && (unsigned)sj < 4u) ? (spT + (si*4 + sj)*64) : zrow;
            const _Float16* arow = wrel + (u*3 + v2)*4096 + (16*mt + m16)*64;
            acc = wmma16(ldsA32(arow, 0,  hi), ldsB32(brow, 0,  hi), acc);
            acc = wmma16(ldsA32(arow, 32, hi), ldsB32(brow, 32, hi), acc);
          }
        #pragma unroll
        for (int r = 0; r < 8; ++r)
          relcT[m16*64 + 16*mt + r + 8*hi] = (_Float16)acc[r];
      }
    }

    // -------- rel_logits: B1[p,(o,s)] = rel_cT x wdT ; rl[n] += qf[o][p] * B1 tile ----
    float rl[4][8];
    #pragma unroll
    for (int n = 0; n < 4; ++n)
      #pragma unroll
      for (int r = 0; r < 8; ++r) rl[n][r] = 0.f;

    {
      // A-fragments of rel_cT are invariant across all 64 o-tiles: hoist to registers.
      const _Float16* arow_rc = relcT + m16*64;
      const v16h a0 = ldsA32(arow_rc, 0,  hi);
      const v16h a1 = ldsA32(arow_rc, 32, hi);
      #pragma unroll
      for (int n = 0; n < 4; ++n) {
        for (int e = 0; e < 16; ++e) {
          const int o = n*16 + e;
          const _Float16* brow = wdt + (o*16 + m16)*64;
          v8f acc = {};
          acc = wmma16(a0, ldsB32(brow, 0,  hi), acc);
          acc = wmma16(a1, ldsB32(brow, 32, hi), acc);
          #pragma unroll
          for (int r = 0; r < 8; ++r)
            rl[n][r] += (float)qf[o*16 + r + 8*hi] * acc[r];
        }
      }
    }

    // -------- per-head MLP logits + softmax + attention --------
    v8f asum = {};
    #pragma unroll
    for (int n = 0; n < 4; ++n) {
      // cat[p][t] = [q5 ; k5]
      {
        const int p = lane >> 1, t0 = (lane & 1) * 16;
        #pragma unroll
        for (int tt = 0; tt < 16; ++tt) {
          const int tc = t0 + tt;
          catb[p*32 + tc] = (tc < 16) ? qf[(n*16 + tc)*16 + p]
                                      : kf[(n*16 + tc - 16)*16 + p];
        }
      }
      // h1 = tanh(cat @ wsf1^T + b1)
      v8f h1 = wmma16(ldsA32(catb + m16*32, 0, hi), ldsB32(wsf1 + m16*32, 0, hi), v8f{});
      const float bias1 = fs[F_BSF1 + m16];
      #pragma unroll
      for (int r = 0; r < 8; ++r)
        h1b[(r + 8*hi)*16 + m16] = (_Float16)tanhf(h1[r] + bias1);

      // logits = h1 @ wsf2^T + b2 + rel_logits
      v8f lg = wmma16(ldsA16(h1b + m16*16, hi), ldsB16(wsf2 + m16*16, hi, zrow), v8f{});
      const float bias2 = fs[F_BSF2 + m16];
      float lv[8];
      #pragma unroll
      for (int r = 0; r < 8; ++r) lv[r] = lg[r] + bias2 + rl[n][r];

      // softmax over s (= 16 lanes within each half-wave)
      #pragma unroll
      for (int r = 0; r < 8; ++r) {
        float mx = lv[r];
        mx = fmaxf(mx, __shfl_xor(mx, 1));
        mx = fmaxf(mx, __shfl_xor(mx, 2));
        mx = fmaxf(mx, __shfl_xor(mx, 4));
        mx = fmaxf(mx, __shfl_xor(mx, 8));
        float ev = __expf(lv[r] - mx);
        float sm = ev;
        sm += __shfl_xor(sm, 1);
        sm += __shfl_xor(sm, 2);
        sm += __shfl_xor(sm, 4);
        sm += __shfl_xor(sm, 8);
        wtile[(r + 8*hi)*16 + m16] = (_Float16)(ev / sm);
      }

      // attn = weights @ v5^T   (K = 16 padded)
      v8f at = wmma16(ldsA16(wtile + m16*16, hi),
                      ldsB16(vmat + (n*16 + m16)*16, hi, zrow), v8f{});
      #pragma unroll
      for (int r = 0; r < 8; ++r) asum[r] += at[r];
    }

    // -------- fold: s_vec[plo*16+d] = sum_{n,phi} attn[n, phi*4+plo, d] --------
    float tacc[4];
    #pragma unroll
    for (int plo = 0; plo < 4; ++plo) {
      float tv = asum[plo] + asum[plo + 4];
      tv += __shfl_xor(tv, 16);
      tacc[plo] = tv;
    }
    if (lane < 16) {
      #pragma unroll
      for (int plo = 0; plo < 4; ++plo) svec[plo*16 + lane] = tacc[plo];
    }

    // -------- output projection: out[o] = svec . wproj[o,:] + 16*bproj[o] --------
    #pragma unroll
    for (int rep = 0; rep < 2; ++rep) {
      const int o = lane + rep*32;
      float acc = 16.f * fs[F_BPROJ + o];
      for (int c = 0; c < 64; ++c)
        acc += svec[c] * (float)wproj[o*64 + c];
      out[(((size_t)bimg*64 + o)*32 + ho)*32 + wo] = acc;
    }
  }
}

extern "C" void kernel_launch(void* const* d_in, const int* in_sizes, int n_in,
                              void* d_out, int out_size, void* d_ws, size_t ws_size,
                              hipStream_t stream) {
  (void)in_sizes; (void)n_in; (void)out_size; (void)d_ws; (void)ws_size;
  const float* x      = (const float*)d_in[0];
  const float* wq     = (const float*)d_in[1];
  const float* wk     = (const float*)d_in[2];
  const float* wv     = (const float*)d_in[3];
  const float* wrel   = (const float*)d_in[4];
  const float* wdec   = (const float*)d_in[5];
  const float* wsf1   = (const float*)d_in[6];
  const float* bsf1   = (const float*)d_in[7];
  const float* wsf2   = (const float*)d_in[8];
  const float* bsf2   = (const float*)d_in[9];
  const float* wproj  = (const float*)d_in[10];
  const float* bproj  = (const float*)d_in[11];
  float* out = (float*)d_out;

  dim3 grid(256), block(128);
  selfattn2d_fused<<<grid, block, SMEM_BYTES, stream>>>(
      x, wq, wk, wv, wrel, wdec, wsf1, bsf1, wsf2, bsf2, wproj, bproj, out);
}
